// SyntonicAttention_1735166787888
// MI455X (gfx1250) — matrix operand order repaired and checked by reading, not compiled, not run on hardware
//
#include <hip/hip_runtime.h>
#include <hip/hip_bf16.h>

#define D_MODEL 1024
#define BATCH   8
#define SEQ     2048
#define MQ      16     // query rows per attention block
#define NKEY    64     // keys per inner block

typedef __attribute__((ext_vector_type(16))) __bf16 v16bf;
typedef __attribute__((ext_vector_type(8)))  float  v8f;
typedef int vi4 __attribute__((__vector_size__(4 * sizeof(int))));

#if __has_builtin(__builtin_amdgcn_global_load_async_to_lds_b128)
#define HAVE_ASYNC_LDS 1
#endif

union Frag {
    v16bf v;
    unsigned short us[16];
    uint4 q[2];
};

__device__ inline unsigned short f2bf(float f) {
    unsigned int u = __float_as_uint(f);
    u += 0x7fffu + ((u >> 16) & 1u);      // round-to-nearest-even
    return (unsigned short)(u >> 16);
}

__device__ inline v8f vzero8() {
    v8f z;
#pragma unroll
    for (int i = 0; i < 8; ++i) z[i] = 0.f;
    return z;
}

__device__ inline v8f wmma_bf16(const Frag& a, const Frag& b, v8f c) {
    return __builtin_amdgcn_wmma_f32_16x16x32_bf16(
        /*neg_a=*/false, a.v, /*neg_b=*/false, b.v,
        /*c_mod=*/(short)0, c, /*reuse_a=*/false, /*reuse_b=*/false);
}

#if defined(HAVE_ASYNC_LDS)
// 16-byte async DMA: global -> LDS, tracked by ASYNCcnt (no VGPR round-trip)
__device__ inline void async_copy16(const void* g, void* l) {
    __builtin_amdgcn_global_load_async_to_lds_b128(
        (vi4*)g, (vi4*)l, 0, 0);
}
__device__ inline void wait_asynccnt0() {
#if __has_builtin(__builtin_amdgcn_s_wait_asynccnt)
    __builtin_amdgcn_s_wait_asynccnt(0);
#else
    asm volatile("s_wait_asynccnt 0x0" ::: "memory");
#endif
}
#endif

// ---------------------------------------------------------------------------
// Kernel A: flash attention (QK^T/sqrt(d) -> softmax -> @V) + entropy stat.
// Writes pre-norm O into d_out rows; atomicAdds per-row entropy into entAcc.
// ---------------------------------------------------------------------------
__global__ __launch_bounds__(256)
void attn_kernel(const float* __restrict__ Q, const float* __restrict__ K,
                 const float* __restrict__ V, float* __restrict__ O,
                 float* __restrict__ entAcc)
{
    extern __shared__ char smem[];
    unsigned short* Qs  = (unsigned short*)smem;                          // [16][1024] bf16
    unsigned short* Vst = (unsigned short*)(smem + 32768);                // [1024][64] bf16 (d-major)
    float*          Sp  = (float*)(smem + 32768 + 131072);                // [2][16][64]
    unsigned short* Ps  = (unsigned short*)(smem + 32768 + 131072 + 8192);// [16][64] bf16
    float* mrow = (float*)(smem + 32768 + 131072 + 8192 + 2048);
    float* lrow = mrow + 16;
    float* trow = lrow + 16;
    float* arow = trow + 16;

    const int tid  = threadIdx.x;
    const int wave = tid >> 5;
    const int lane = tid & 31;
    const int hi16 = lane >> 4;
    const int l15  = lane & 15;

    const int b  = blockIdx.x >> 7;            // SEQ/MQ = 128 tiles per batch
    const int q0 = (blockIdx.x & 127) * MQ;
    const float scale = 0.03125f;              // 1/sqrt(1024)

    const float* Qb = Q + ((size_t)(b * SEQ + q0)) * D_MODEL;
    const float* Kb = K + ((size_t)b * SEQ) * D_MODEL;
    const float* Vb = V + ((size_t)b * SEQ) * D_MODEL;

    // ---- stage Q tile to bf16 LDS ----
#if defined(HAVE_ASYNC_LDS)
    {
        // async-DMA the raw fp32 Q tile into LDS (overlays Vst, used later),
        // then convert LDS->LDS to bf16.
        float* Qstage = (float*)(smem + 32768);
        for (int i = tid; i < 4096; i += 256)          // 4096 x 16B chunks
            async_copy16(Qb + (size_t)i * 4, Qstage + (size_t)i * 4);
        wait_asynccnt0();
        __syncthreads();
        for (int i = tid; i < MQ * 256; i += 256) {
            int r = i >> 8, c4 = (i & 255) << 2;
            float4 f = *(const float4*)(Qstage + r * D_MODEL + c4);
            unsigned int lo = (unsigned int)f2bf(f.x) | ((unsigned int)f2bf(f.y) << 16);
            unsigned int hi = (unsigned int)f2bf(f.z) | ((unsigned int)f2bf(f.w) << 16);
            *(uint2*)(Qs + r * D_MODEL + c4) = make_uint2(lo, hi);
        }
    }
#else
    for (int i = tid; i < MQ * 256; i += 256) {
        int r = i >> 8, c4 = (i & 255) << 2;
        float4 f = *(const float4*)(Qb + (size_t)r * D_MODEL + c4);
        unsigned int lo = (unsigned int)f2bf(f.x) | ((unsigned int)f2bf(f.y) << 16);
        unsigned int hi = (unsigned int)f2bf(f.z) | ((unsigned int)f2bf(f.w) << 16);
        *(uint2*)(Qs + r * D_MODEL + c4) = make_uint2(lo, hi);
    }
#endif
    if (tid < 16) { mrow[tid] = -INFINITY; lrow[tid] = 0.f; trow[tid] = 0.f; }

    v8f acc[8];
#pragma unroll
    for (int t = 0; t < 8; ++t) acc[t] = vzero8();
    __syncthreads();

    const int jsub = wave & 3;   // which 16-key score subtile
    const int dh   = wave >> 2;  // which d-half (0: d<512, 1: d>=512)

    for (int kb = 0; kb < SEQ; kb += NKEY) {
        // ---- Phase A: partial scores S[j] over this wave's d-half ----
        v8f sc = vzero8();
        const float* Kblk =
            Kb + (size_t)(kb + jsub * 16 + l15) * D_MODEL + dh * 512 + hi16 * 16;
        __builtin_prefetch(Kblk + NKEY * D_MODEL, 0, 0);  // next key block
#pragma unroll 4
        for (int dk = 0; dk < 512; dk += 32) {
            const int k0 = dh * 512 + dk;
            Frag a, bf;
            a.q[0] = *(const uint4*)(Qs + l15 * D_MODEL + k0 + hi16 * 16);
            a.q[1] = *(const uint4*)(Qs + l15 * D_MODEL + k0 + hi16 * 16 + 8);
            float4 k0v = *(const float4*)(Kblk + dk);
            float4 k1v = *(const float4*)(Kblk + dk + 4);
            float4 k2v = *(const float4*)(Kblk + dk + 8);
            float4 k3v = *(const float4*)(Kblk + dk + 12);
            bf.us[0]  = f2bf(k0v.x); bf.us[1]  = f2bf(k0v.y);
            bf.us[2]  = f2bf(k0v.z); bf.us[3]  = f2bf(k0v.w);
            bf.us[4]  = f2bf(k1v.x); bf.us[5]  = f2bf(k1v.y);
            bf.us[6]  = f2bf(k1v.z); bf.us[7]  = f2bf(k1v.w);
            bf.us[8]  = f2bf(k2v.x); bf.us[9]  = f2bf(k2v.y);
            bf.us[10] = f2bf(k2v.z); bf.us[11] = f2bf(k2v.w);
            bf.us[12] = f2bf(k3v.x); bf.us[13] = f2bf(k3v.y);
            bf.us[14] = f2bf(k3v.z); bf.us[15] = f2bf(k3v.w);
            sc = wmma_bf16(a, bf, sc);
        }
#pragma unroll
        for (int i = 0; i < 8; ++i) {
            int row = i + hi16 * 8;
            Sp[(dh * 16 + row) * 64 + jsub * 16 + l15] = sc[i];
        }

        // ---- stage V block transposed: Vst[d][k] bf16 ----
        for (int i = tid; i < NKEY * 256; i += 256) {
            int k = i >> 8, c4 = (i & 255) << 2;
            float4 f = *(const float4*)(Vb + (size_t)(kb + k) * D_MODEL + c4);
            Vst[(c4 + 0) * NKEY + k] = f2bf(f.x);
            Vst[(c4 + 1) * NKEY + k] = f2bf(f.y);
            Vst[(c4 + 2) * NKEY + k] = f2bf(f.z);
            Vst[(c4 + 3) * NKEY + k] = f2bf(f.w);
        }
        __syncthreads();

        // ---- online softmax (16 threads per row, 4 scores each) ----
        {
            int row = tid >> 4, c0 = (tid & 15) << 2;
            float s[4];
            float bm = -INFINITY;
#pragma unroll
            for (int j = 0; j < 4; ++j) {
                s[j] = (Sp[row * 64 + c0 + j] + Sp[(16 + row) * 64 + c0 + j]) * scale;
                bm = fmaxf(bm, s[j]);
            }
#pragma unroll
            for (int m = 8; m >= 1; m >>= 1) bm = fmaxf(bm, __shfl_xor(bm, m, 16));
            float mo = mrow[row];
            float mn = fmaxf(mo, bm);
            float al = __expf(mo - mn);
            float ps = 0.f, ts = 0.f;
            unsigned short pb[4];
#pragma unroll
            for (int j = 0; j < 4; ++j) {
                float p = __expf(s[j] - mn);
                ps += p;
                ts += p * s[j];
                pb[j] = f2bf(p);
            }
#pragma unroll
            for (int m = 8; m >= 1; m >>= 1) {
                ps += __shfl_xor(ps, m, 16);
                ts += __shfl_xor(ts, m, 16);
            }
            if ((tid & 15) == 0) {
                mrow[row] = mn;
                lrow[row] = al * lrow[row] + ps;
                trow[row] = al * trow[row] + ts;
                arow[row] = al;
            }
            unsigned int lo = (unsigned int)pb[0] | ((unsigned int)pb[1] << 16);
            unsigned int hi = (unsigned int)pb[2] | ((unsigned int)pb[3] << 16);
            *(uint2*)(Ps + row * 64 + c0) = make_uint2(lo, hi);
        }
        __syncthreads();

        // ---- Phase C: rescale O accumulators and accumulate P @ V ----
        {
            float4 a0 = *(const float4*)(arow + hi16 * 8);
            float4 a1 = *(const float4*)(arow + hi16 * 8 + 4);
            float av[8] = {a0.x, a0.y, a0.z, a0.w, a1.x, a1.y, a1.z, a1.w};
#pragma unroll
            for (int t = 0; t < 8; ++t)
#pragma unroll
                for (int i = 0; i < 8; ++i) acc[t][i] *= av[i];

            Frag pa0, pa1;
            pa0.q[0] = *(const uint4*)(Ps + l15 * 64 + hi16 * 16);
            pa0.q[1] = *(const uint4*)(Ps + l15 * 64 + hi16 * 16 + 8);
            pa1.q[0] = *(const uint4*)(Ps + l15 * 64 + 32 + hi16 * 16);
            pa1.q[1] = *(const uint4*)(Ps + l15 * 64 + 32 + hi16 * 16 + 8);
            const int n0w = wave * 128;
#pragma unroll
            for (int ct = 0; ct < 8; ++ct) {
                const int ncol = n0w + ct * 16 + l15;
                Frag b0, b1;
                const uint4* vp0 = (const uint4*)(Vst + ncol * NKEY + hi16 * 16);
                const uint4* vp1 = (const uint4*)(Vst + ncol * NKEY + 32 + hi16 * 16);
                b0.q[0] = vp0[0]; b0.q[1] = vp0[1];
                b1.q[0] = vp1[0]; b1.q[1] = vp1[1];
                acc[ct] = wmma_bf16(pa0, b0, acc[ct]);
                acc[ct] = wmma_bf16(pa1, b1, acc[ct]);
            }
        }
        __syncthreads();
    }

    // ---- finalize: O = acc / l ----
    {
        float4 l0 = *(const float4*)(lrow + hi16 * 8);
        float4 l1 = *(const float4*)(lrow + hi16 * 8 + 4);
        float lv[8] = {l0.x, l0.y, l0.z, l0.w, l1.x, l1.y, l1.z, l1.w};
        float* Ob = O + (size_t)(b * SEQ + q0) * D_MODEL;
#pragma unroll
        for (int ct = 0; ct < 8; ++ct) {
            int col = wave * 128 + ct * 16 + l15;
#pragma unroll
            for (int i = 0; i < 8; ++i) {
                int row = i + hi16 * 8;
                Ob[(size_t)row * D_MODEL + col] = acc[ct][i] / lv[i];
            }
        }
    }
    if (tid < 16) {
        float H = mrow[tid] + __logf(lrow[tid]) - trow[tid] / lrow[tid];
        atomicAdd(entAcc, H);
    }
}

// ---------------------------------------------------------------------------
// Kernel B: h = O @ W + b, then LayerNorm(gamma,beta), in place on d_out.
// 64 rows per block; O rows staged bf16 in LDS so global overwrite is safe.
// W tiles async-DMA'd into LDS, then converted/transposed to bf16.
// ---------------------------------------------------------------------------
__global__ __launch_bounds__(256)
void harm_kernel(float* __restrict__ OutIO, const float* __restrict__ Wm,
                 const float* __restrict__ bias, const float* __restrict__ gamma,
                 const float* __restrict__ beta)
{
    extern __shared__ char smem[];
    unsigned short* Os   = (unsigned short*)smem;             // [64][1024] bf16
    unsigned short* Wt   = (unsigned short*)(smem + 131072);  // [128][32] bf16 (n-major)
    float*          Wraw = (float*)(smem + 139264);           // [32][128] fp32 staging
    float* rsum  = (float*)(smem + 155648);
    float* rsq   = rsum + 64;
    float* mus   = rsq + 64;
    float* rstds = mus + 64;

    const int tid  = threadIdx.x;
    const int wave = tid >> 5;
    const int lane = tid & 31;
    const int hi16 = lane >> 4;
    const int l15  = lane & 15;

    const size_t r0 = (size_t)blockIdx.x * 64;
    float* Orows = OutIO + r0 * D_MODEL;

    // stage O rows -> bf16 LDS
    for (int i = tid; i < 64 * 256; i += 256) {
        int r = i >> 8, c4 = (i & 255) << 2;
        float4 f = *(const float4*)(Orows + (size_t)r * D_MODEL + c4);
        unsigned int lo = (unsigned int)f2bf(f.x) | ((unsigned int)f2bf(f.y) << 16);
        unsigned int hi = (unsigned int)f2bf(f.z) | ((unsigned int)f2bf(f.w) << 16);
        *(uint2*)(Os + r * D_MODEL + c4) = make_uint2(lo, hi);
    }
    if (tid < 64) { rsum[tid] = 0.f; rsq[tid] = 0.f; }
    __syncthreads();

    const int rt = wave >> 1;  // row tile (16 rows)
    const int ch = wave & 1;   // column half of the 128-wide n-block

    for (int nb = 0; nb < D_MODEL; nb += 128) {
        v8f accb[4];
#pragma unroll
        for (int t = 0; t < 4; ++t) accb[t] = vzero8();

        for (int kb = 0; kb < D_MODEL; kb += 32) {
            __syncthreads();
            // stage W[kb..kb+32)[nb..nb+128) transposed (n-major) in bf16
#if defined(HAVE_ASYNC_LDS)
            for (int i = tid; i < 1024; i += 256) {   // 1024 x 16B chunks
                int kr = i >> 5, c4 = (i & 31) << 2;
                async_copy16(Wm + (size_t)(kb + kr) * D_MODEL + nb + c4,
                             Wraw + kr * 128 + c4);
            }
            wait_asynccnt0();
            __syncthreads();
            for (int i = tid; i < 1024; i += 256) {
                int kr = i >> 5, c4 = (i & 31) << 2;
                float4 f = *(const float4*)(Wraw + kr * 128 + c4);
                Wt[(c4 + 0) * 32 + kr] = f2bf(f.x);
                Wt[(c4 + 1) * 32 + kr] = f2bf(f.y);
                Wt[(c4 + 2) * 32 + kr] = f2bf(f.z);
                Wt[(c4 + 3) * 32 + kr] = f2bf(f.w);
            }
#else
            for (int i = tid; i < 1024; i += 256) {
                int kr = i >> 5, c4 = (i & 31) << 2;
                float4 f = *(const float4*)(Wm + (size_t)(kb + kr) * D_MODEL + nb + c4);
                Wt[(c4 + 0) * 32 + kr] = f2bf(f.x);
                Wt[(c4 + 1) * 32 + kr] = f2bf(f.y);
                Wt[(c4 + 2) * 32 + kr] = f2bf(f.z);
                Wt[(c4 + 3) * 32 + kr] = f2bf(f.w);
            }
#endif
            __syncthreads();

            Frag a;
            a.q[0] = *(const uint4*)(Os + (rt * 16 + l15) * D_MODEL + kb + hi16 * 16);
            a.q[1] = *(const uint4*)(Os + (rt * 16 + l15) * D_MODEL + kb + hi16 * 16 + 8);
#pragma unroll
            for (int ct = 0; ct < 4; ++ct) {
                Frag bfr;
                const uint4* wp =
                    (const uint4*)(Wt + (ch * 64 + ct * 16 + l15) * 32 + hi16 * 16);
                bfr.q[0] = wp[0];
                bfr.q[1] = wp[1];
                accb[ct] = wmma_bf16(a, bfr, accb[ct]);
            }
        }

        // epilogue: bias, store h, accumulate layernorm stats
#pragma unroll
        for (int ct = 0; ct < 4; ++ct) {
            int col = nb + ch * 64 + ct * 16 + l15;
            float bv = bias[col];
#pragma unroll
            for (int i = 0; i < 8; ++i) {
                int row = rt * 16 + i + hi16 * 8;
                float h = accb[ct][i] + bv;
                Orows[(size_t)row * D_MODEL + col] = h;
                atomicAdd(&rsum[row], h);
                atomicAdd(&rsq[row], h * h);
            }
        }
    }
    __syncthreads();

    if (tid < 64) {
        float mu  = rsum[tid] * (1.f / 1024.f);
        float var = rsq[tid] * (1.f / 1024.f) - mu * mu;
        mus[tid]   = mu;
        rstds[tid] = rsqrtf(var + 1e-5f);
    }
    __syncthreads();

    // pass 2: normalize in place
    for (int i = tid; i < 64 * 256; i += 256) {
        int r = i >> 8, c4 = (i & 255) << 2;
        float4 h  = *(const float4*)(Orows + (size_t)r * D_MODEL + c4);
        float4 g  = *(const float4*)(gamma + c4);
        float4 be = *(const float4*)(beta + c4);
        float mu = mus[r], rs = rstds[r];
        h.x = (h.x - mu) * rs * g.x + be.x;
        h.y = (h.y - mu) * rs * g.y + be.y;
        h.z = (h.z - mu) * rs * g.z + be.z;
        h.w = (h.w - mu) * rs * g.w + be.w;
        *(float4*)(Orows + (size_t)r * D_MODEL + c4) = h;
    }
}

// ---------------------------------------------------------------------------
// Kernel C: syntony scalar
// ---------------------------------------------------------------------------
__global__ void syntony_kernel(const float* __restrict__ entAcc,
                               float* __restrict__ out)
{
    float me = entAcc[0] / (float)(BATCH * SEQ);
    float s  = 1.f - me / logf((float)SEQ);
    out[0]   = fminf(fmaxf(s, 0.f), 1.f);
}

extern "C" void kernel_launch(void* const* d_in, const int* in_sizes, int n_in,
                              void* d_out, int out_size, void* d_ws, size_t ws_size,
                              hipStream_t stream)
{
    (void)in_sizes; (void)n_in; (void)out_size; (void)ws_size;
    const float* Q  = (const float*)d_in[0];
    const float* K  = (const float*)d_in[1];
    const float* V  = (const float*)d_in[2];
    const float* W  = (const float*)d_in[3];
    const float* hb = (const float*)d_in[4];
    const float* g  = (const float*)d_in[5];
    const float* be = (const float*)d_in[6];
    float* out = (float*)d_out;
    float* ent = (float*)d_ws;

    (void)hipMemsetAsync(ent, 0, sizeof(float), stream);

    size_t smemA = 32768 + 131072 + 8192 + 2048 + 4 * 16 * sizeof(float);
    attn_kernel<<<dim3(BATCH * (SEQ / MQ)), dim3(256), smemA, stream>>>(Q, K, V, out, ent);

    syntony_kernel<<<dim3(1), dim3(1), 0, stream>>>(ent, out + (size_t)BATCH * SEQ * D_MODEL);

    size_t smemB = 155648 + 4 * 64 * sizeof(float);
    harm_kernel<<<dim3((BATCH * SEQ) / 64), dim3(256), smemB, stream>>>(out, W, hb, g, be);
}
